// BatchRecurrentAttention16Layer_40501541601427
// MI455X (gfx1250) — matrix operand order repaired and checked by reading, not compiled, not run on hardware
//
#include <hip/hip_runtime.h>

typedef __attribute__((ext_vector_type(16))) _Float16 v16h;
typedef __attribute__((ext_vector_type(8)))  float    v8f;
typedef unsigned int u32x4 __attribute__((ext_vector_type(4)));
typedef int          i32x4 __attribute__((ext_vector_type(4)));
typedef int          i32x8 __attribute__((ext_vector_type(8)));

#define Mm 8
#define Bb 64
#define Ss 1024
#define Kk 256
#define Hh 8
#define Ff 1024

#if __has_builtin(__builtin_amdgcn_tensor_load_to_lds)
#define HAVE_TDM 1
#else
#define HAVE_TDM 0
#endif

__device__ __forceinline__ v8f wmma16(v16h a, v16h b, v8f c) {
  return __builtin_amdgcn_wmma_f32_16x16x32_f16(false, a, false, b, (short)0, c, false, false);
}

// A-fragment (16x32 f16): lane<16 -> row r0+lane, K {k0..k0+7, k0+16..k0+23};
// lane>=16 -> row r0+lane-16, +8 on K. Caller passes p = base + row*ld + k0 + ((lane>>4)<<3).
__device__ __forceinline__ v16h ldfragA_h(const _Float16* p) {
  v16h f;
#pragma unroll
  for (int i = 0; i < 8; ++i) { f[i] = p[i]; f[i + 8] = p[i + 16]; }
  return f;
}
__device__ __forceinline__ v16h ldfragA_f(const float* p) {
  v16h f;
#pragma unroll
  for (int i = 0; i < 8; ++i) { f[i] = (_Float16)p[i]; f[i + 8] = (_Float16)p[i + 16]; }
  return f;
}
// B-fragment (32x16 f16): lane holds column n=lane&15; lanes<16 -> K {k0..k0+7, k0+16..k0+23}, lanes>=16 -> +8 on K.
// Caller passes p = baseT + n*ld + k0 + ((lane>>4)<<3).
__device__ __forceinline__ v16h ldfragB_h(const _Float16* p) {
  v16h f;
#pragma unroll
  for (int i = 0; i < 8; ++i) { f[i] = p[i]; f[i + 8] = p[i + 16]; }
  return f;
}

// ---------------- LDS-tiled transpose + fp32 -> f16: dst[g][c][r] = src[g][r][c] ----------------
// R, C multiples of 32. Coalesced reads and writes.
__global__ void k_tcvt(const float* __restrict__ src, _Float16* __restrict__ dst, int R, int C) {
  __shared__ _Float16 tile[32][33];
  int g = blockIdx.z;
  int r0 = blockIdx.x * 32;
  int c0 = blockIdx.y * 32;
  int c = threadIdx.x & 31, rr = threadIdx.x >> 5;
  const float* S = src + (size_t)g * R * C;
  _Float16* D = dst + (size_t)g * R * C;
  for (int r = rr; r < 32; r += 8)
    tile[r][c] = (_Float16)S[(size_t)(r0 + r) * C + c0 + c];
  __syncthreads();
  for (int r = rr; r < 32; r += 8)
    D[(size_t)(c0 + r) * R + r0 + c] = tile[c][r];
}

// ---------------- key (S,B,K) f32 -> keyh (B,S,K) f16 (coalesced both sides) ----------------
__global__ void k_cvtkey(const float* __restrict__ key, _Float16* __restrict__ keyh) {
  size_t idx = (size_t)blockIdx.x * 256 + threadIdx.x; // B*S*K
  int k = (int)(idx & 255);
  int s = (int)((idx >> 8) & 1023);
  int b = (int)(idx >> 18);
  keyh[idx] = (_Float16)key[((size_t)s * Bb + b) * Kk + k];
}

// ---------------- value (S,B,K) f32 -> valT (B,K,S) f16 via LDS tile transpose ----------------
__global__ void k_tval(const float* __restrict__ value, _Float16* __restrict__ valT) {
  __shared__ _Float16 tile[32][33];
  int b = blockIdx.z;
  int s0 = blockIdx.x * 32;
  int k0 = blockIdx.y * 32;
  int c = threadIdx.x & 31, r0 = threadIdx.x >> 5;
  for (int r = r0; r < 32; r += 8)
    tile[r][c] = (_Float16)value[((size_t)(s0 + r) * Bb + b) * Kk + k0 + c];
  __syncthreads();
  for (int r = r0; r < 32; r += 8)
    valT[((size_t)b * Kk + k0 + r) * Ss + s0 + c] = tile[c][r];
}

// ---------------- q projection: q = q0 @ wq + bq  (per m: 64x256 @ 256x256) ----------------
__global__ __launch_bounds__(256) void k_qproj(const float* __restrict__ q0,
                                               const _Float16* __restrict__ wqT,
                                               const float* __restrict__ bq,
                                               float* __restrict__ qout) {
  int m = blockIdx.x;
  int lane = threadIdx.x & 31, w = threadIdx.x >> 5;
  const float* A = q0 + (size_t)m * Bb * Kk;
  const _Float16* Bt = wqT + (size_t)m * Kk * Kk;
  for (int t = w; t < 64; t += 8) {
    int r0 = (t >> 4) * 16, n0 = (t & 15) * 16;
    v8f c = {};
    for (int kk = 0; kk < Kk; kk += 32) {
      v16h a = ldfragA_f(A + (size_t)(r0 + (lane & 15)) * Kk + kk + ((lane >> 4) << 3));
      v16h b = ldfragB_h(Bt + (size_t)(n0 + (lane & 15)) * Kk + kk + ((lane >> 4) << 3));
      c = wmma16(a, b, c);
    }
    int col = n0 + (lane & 15);
    int rbase = r0 + ((lane >> 4) << 3);
#pragma unroll
    for (int r = 0; r < 8; ++r)
      qout[((size_t)m * Bb + rbase + r) * Kk + col] = c[r] + bq[m * Kk + col];
  }
}

// ---------------- fold wk with q, packing TWO modules per 16-row tile ----------------
// P[pair,b,row,k], row = (m&1)*8 + h, m = pair*2 + (row>>3):
//   P = sum_{d in head h} wk[m,k,d] * q[m,b,d]
__global__ void k_pfold(const float* __restrict__ wk, const float* __restrict__ q,
                        _Float16* __restrict__ P) {
  size_t idx = (size_t)blockIdx.x * 256 + threadIdx.x; // 4*B*16*256
  int k = (int)(idx & 255);
  int row = (int)((idx >> 8) & 15);
  int b = (int)((idx >> 12) & 63);
  int pair = (int)(idx >> 18);
  int m = pair * 2 + (row >> 3);
  int h = row & 7;
  const float* wr = wk + ((size_t)m * Kk + k) * Kk + h * 32;
  const float* qr = q + ((size_t)m * Bb + b) * Kk + h * 32;
  float acc = 0.f;
#pragma unroll 8
  for (int d = 0; d < 32; ++d) acc += wr[d] * qr[d];
  P[idx] = (_Float16)acc;
}

// ---------------- scores: per (b, m-pair): 1024x256 @ 256x16 (16 cols = 2 modules x 8 heads) ----------------
// P tile (8 KB) staged into LDS via TDM (tensor_load_to_lds) when available.
__global__ __launch_bounds__(256) void k_scores(const _Float16* __restrict__ keyh,
                                                const _Float16* __restrict__ P,
                                                float* __restrict__ sc) {
  int b = blockIdx.x, pair = blockIdx.y;
  int lane = threadIdx.x & 31, w = threadIdx.x >> 5;
  __shared__ __align__(16) _Float16 sP[16 * Kk]; // 8 KB
  const _Float16* Pg = P + (size_t)((pair * Bb + b) * 16) * Kk;
#if HAVE_TDM
  if (threadIdx.x < 32) {
    unsigned long long ga = (unsigned long long)(uintptr_t)Pg;
    unsigned lds = (unsigned)(uintptr_t)&sP[0];
    // D# group0: count=1 | lds_addr | global_addr[56:0] | type=2
    u32x4 g0 = {1u, lds, (unsigned)ga, (unsigned)((ga >> 32) & 0x1FFFFFFu) | (2u << 30)};
    // D# group1: data_size=8B; tensor 64x16 (8B units), tile 64x16, dim0_stride=64
    i32x8 g1 = {(3 << 16), (64 << 16), (16 << 16), (64 << 16), 16, 64, 0, 0};
    i32x4 z4 = {0, 0, 0, 0};
#if __has_include(<hip/amd_detail/amd_gfx1250_TDM.h>)
    i32x8 z8 = {0, 0, 0, 0, 0, 0, 0, 0};
    __builtin_amdgcn_tensor_load_to_lds(g0, g1, z4, z4, z8, 0);
#else
    __builtin_amdgcn_tensor_load_to_lds(g0, g1, z4, z4, 0);
#endif
    __builtin_amdgcn_s_wait_tensorcnt(0);
  }
#else
  {
    const uint4* s4 = (const uint4*)Pg;
    uint4* d4 = (uint4*)sP;
    for (int i = threadIdx.x; i < 16 * Kk / 8; i += 256) d4[i] = s4[i];
  }
#endif
  __syncthreads();

  int n = lane & 15;
  int m = pair * 2 + (n >> 3);
  int h = n & 7;
  float* outbase = sc + (size_t)((m * Bb + b) * Hh + h) * Ss;
  for (int rt = w; rt < 64; rt += 8) {
    int s0 = rt * 16;
    const _Float16* Arow =
        keyh + ((size_t)b * Ss + s0 + (lane & 15)) * Kk + ((lane >> 4) << 3);
    const _Float16* Brow = sP + (size_t)(lane & 15) * Kk + ((lane >> 4) << 3);
    v8f c = {};
    for (int kk = 0; kk < Kk; kk += 32) {
      v16h a = ldfragA_h(Arow + kk);
      v16h bb = ldfragB_h(Brow + kk);
      c = wmma16(a, bb, c);
    }
    int sbase = s0 + ((lane >> 4) << 3);
#pragma unroll
    for (int r = 0; r < 8; ++r) outbase[sbase + r] = c[r] * 0.17677669529663689f;
  }
}

// ---------------- softmax over s (in place) + write packed f16 aw (A matrix for Y GEMM) ----------------
// awp[pair,b,row,s] with row = (m&1)*8 + h.
__global__ void k_softmax(float* __restrict__ sc, _Float16* __restrict__ awp) {
  int h = blockIdx.x, b = blockIdx.y, m = blockIdx.z;
  int t = threadIdx.x;
  _Float16* prow = awp + (size_t)(((m >> 1) * Bb + b) * 16 + ((m & 1) << 3) + h) * Ss;
  float* row = sc + (size_t)((m * Bb + b) * Hh + h) * Ss;
  __shared__ float red[256];
  float v[4], mx = -1e30f;
#pragma unroll
  for (int i = 0; i < 4; ++i) { v[i] = row[t + i * 256]; mx = fmaxf(mx, v[i]); }
  red[t] = mx; __syncthreads();
  for (int s = 128; s > 0; s >>= 1) { if (t < s) red[t] = fmaxf(red[t], red[t + s]); __syncthreads(); }
  mx = red[0]; __syncthreads();
  float sum = 0.f;
#pragma unroll
  for (int i = 0; i < 4; ++i) { v[i] = __expf(v[i] - mx); sum += v[i]; }
  red[t] = sum; __syncthreads();
  for (int s = 128; s > 0; s >>= 1) { if (t < s) red[t] += red[t + s]; __syncthreads(); }
  float inv = 1.0f / red[0];
#pragma unroll
  for (int i = 0; i < 4; ++i) {
    float a = v[i] * inv;
    row[t + i * 256] = a;
    prow[t + i * 256] = (_Float16)a;
  }
}

// ---------------- attn_weights = mean over heads of aw ----------------
__global__ void k_awmean(const float* __restrict__ sc, float* __restrict__ awo) {
  size_t idx = (size_t)blockIdx.x * 256 + threadIdx.x; // M*B*S
  int s = (int)(idx & 1023);
  size_t mb = idx >> 10;
  const float* p = sc + mb * Hh * Ss + s;
  float acc = 0.f;
#pragma unroll
  for (int h = 0; h < Hh; ++h) acc += p[(size_t)h * Ss];
  awo[idx] = acc * 0.125f;
}

// ---------------- Y: per (b, m-pair): 16x1024 @ 1024x256 (16 rows = 2 modules x 8 heads) ----------------
// aw tile (32 KB) staged cooperatively into LDS; valT B-fragments are two 16B loads per lane, 32B apart.
__global__ __launch_bounds__(256) void k_y(const _Float16* __restrict__ valT,
                                           const _Float16* __restrict__ awp,
                                           float* __restrict__ Y) {
  int b = blockIdx.x, pair = blockIdx.y;
  int lane = threadIdx.x & 31, w = threadIdx.x >> 5;
  __shared__ __align__(16) _Float16 sA[16 * Ss]; // 32 KB
  {
    const uint4* s4 = (const uint4*)(awp + (size_t)((pair * Bb + b) * 16) * Ss);
    uint4* d4 = (uint4*)sA;
    for (int i = threadIdx.x; i < 16 * Ss / 8; i += 256) d4[i] = s4[i];
  }
  __syncthreads();
  int mloc = pair * 2 + (lane >> 4); // lanes<16 hold C rows 0..7 (m even), lanes>=16 rows 8..15 (m odd)
  for (int ct = w * 2; ct < w * 2 + 2; ++ct) {
    int n0 = ct * 16;
    int vk = n0 + (lane & 15);
    const _Float16* Arow = sA + (size_t)(lane & 15) * Ss + ((lane >> 4) << 3);
    const _Float16* Brow = valT + ((size_t)b * Kk + vk) * Ss + ((lane >> 4) << 3);
    v8f c = {};
    for (int kk = 0; kk < Ss; kk += 32) {
      __builtin_prefetch(Brow + kk + 512, 0, 0);
      v16h a = ldfragA_h(Arow + kk);
      v16h bb = ldfragB_h(Brow + kk);
      c = wmma16(a, bb, c);
    }
    float* outp = Y + (size_t)(mloc * Bb + b) * Hh * Kk + vk; // row r -> head h=r
#pragma unroll
    for (int r = 0; r < 8; ++r) outp[(size_t)r * Kk] = c[r];
  }
}

// ---------------- ctx via WMMA: per (m,h): (64 x 256) @ (256 x 32), +bv -> ctxh f16 ----------------
// A = Y[m, b, h, :] (row b, row-stride H*K, fp32 converted in loader); B = wvT rows do in [h*32, h*32+32).
__global__ __launch_bounds__(256) void k_ctx(const float* __restrict__ Y,
                                             const _Float16* __restrict__ wvT,
                                             const float* __restrict__ bv,
                                             _Float16* __restrict__ ctxh) {
  int h = blockIdx.x, m = blockIdx.y;
  int lane = threadIdx.x & 31, w = threadIdx.x >> 5;
  const float* A = Y + (size_t)(m * Bb) * Hh * Kk + (size_t)h * Kk; // row b, ld = H*K
  const _Float16* Bt = wvT + (size_t)m * Kk * Kk;                   // rows do, ld = 256
  // 8 tiles: 4 row tiles (b) x 2 col tiles (do within head h); one per wave
  int rt = w >> 1, ctile = w & 1;
  int r0 = rt * 16;
  int n0 = h * 32 + ctile * 16;
  v8f c = {};
  for (int kk = 0; kk < Kk; kk += 32) {
    v16h a = ldfragA_f(A + (size_t)(r0 + (lane & 15)) * (Hh * Kk) + kk + ((lane >> 4) << 3));
    v16h bb = ldfragB_h(Bt + (size_t)(n0 + (lane & 15)) * Kk + kk + ((lane >> 4) << 3));
    c = wmma16(a, bb, c);
  }
  int col = n0 + (lane & 15);
  int rbase = r0 + ((lane >> 4) << 3);
  float bias = bv[m * Kk + col];
#pragma unroll
  for (int r = 0; r < 8; ++r)
    ctxh[((size_t)m * Bb + rbase + r) * Kk + col] = (_Float16)(c[r] + bias);
}

// ---------------- attn_out = ctx @ wo + bo ; also fc_in[:, :256] = relu(attn_out) ----------------
__global__ __launch_bounds__(256) void k_attn_out(const _Float16* __restrict__ ctxh,
                                                  const _Float16* __restrict__ woT,
                                                  const float* __restrict__ bo,
                                                  float* __restrict__ outa,
                                                  _Float16* __restrict__ fcin) {
  int m = blockIdx.x;
  int lane = threadIdx.x & 31, w = threadIdx.x >> 5;
  const _Float16* A = ctxh + (size_t)m * Bb * Kk;
  const _Float16* Bt = woT + (size_t)m * Kk * Kk;
  for (int t = w; t < 64; t += 8) {
    int r0 = (t >> 4) * 16, n0 = (t & 15) * 16;
    v8f c = {};
    for (int kk = 0; kk < Kk; kk += 32) {
      v16h a = ldfragA_h(A + (size_t)(r0 + (lane & 15)) * Kk + kk + ((lane >> 4) << 3));
      v16h bb = ldfragB_h(Bt + (size_t)(n0 + (lane & 15)) * Kk + kk + ((lane >> 4) << 3));
      c = wmma16(a, bb, c);
    }
    int col = n0 + (lane & 15);
    int rbase = r0 + ((lane >> 4) << 3);
#pragma unroll
    for (int r = 0; r < 8; ++r) {
      float vv = c[r] + bo[m * Kk + col];
      size_t row = (size_t)m * Bb + rbase + r;
      outa[row * Kk + col] = vv;
      fcin[row * 512 + col] = (_Float16)fmaxf(vv, 0.f);
    }
  }
}

// ---------------- fc_in[:, 256:512] = relu(s0) ----------------
__global__ void k_fcin_s0(const float* __restrict__ s0, _Float16* __restrict__ fcin) {
  size_t idx = (size_t)blockIdx.x * 256 + threadIdx.x; // M*B*256
  int cc = (int)(idx & 255);
  size_t mb = idx >> 8;
  fcin[mb * 512 + 256 + cc] = (_Float16)fmaxf(s0[idx], 0.f);
}

// ---------------- MLP layer 1 (8 heads: q,k,v,s + 4 gates): hid = relu(fc_in @ w1 + b1) ----------------
__global__ __launch_bounds__(256) void k_mlp1(const _Float16* __restrict__ fcin,
                                              const _Float16* w1q, const _Float16* w1k,
                                              const _Float16* w1v, const _Float16* w1s,
                                              const _Float16* gw1,
                                              const float* b1q, const float* b1k,
                                              const float* b1v, const float* b1s,
                                              const float* gb1,
                                              _Float16* __restrict__ hid) {
  int cg = blockIdx.x, m = blockIdx.y, head = blockIdx.z;
  const _Float16* Wt; const float* bias;
  switch (head) {
    case 0: Wt = w1q; bias = b1q; break;
    case 1: Wt = w1k; bias = b1k; break;
    case 2: Wt = w1v; bias = b1v; break;
    case 3: Wt = w1s; bias = b1s; break;
    default: Wt = gw1 + (size_t)(head - 4) * Mm * Ff * 512;
             bias = gb1 + (size_t)(head - 4) * Mm * Ff; break;
  }
  Wt += (size_t)m * Ff * 512; // rows f (1024), ld 512
  bias += (size_t)m * Ff;
  const _Float16* A = fcin + (size_t)m * Bb * 512;
  _Float16* outp = hid + (size_t)(head * Mm + m) * Bb * Ff;
  int lane = threadIdx.x & 31, w = threadIdx.x >> 5;
  for (int t = w; t < 64; t += 8) {
    int r0 = (t >> 4) * 16;
    int n0 = cg * 256 + (t & 15) * 16;
    v8f c = {};
    for (int kk = 0; kk < 512; kk += 32) {
      v16h a = ldfragA_h(A + (size_t)(r0 + (lane & 15)) * 512 + kk + ((lane >> 4) << 3));
      v16h bb = ldfragB_h(Wt + (size_t)(n0 + (lane & 15)) * 512 + kk + ((lane >> 4) << 3));
      c = wmma16(a, bb, c);
    }
    int col = n0 + (lane & 15);
    int rbase = r0 + ((lane >> 4) << 3);
#pragma unroll
    for (int r = 0; r < 8; ++r)
      outp[(size_t)(rbase + r) * Ff + col] = (_Float16)fmaxf(c[r] + bias[col], 0.f);
  }
}

// ---------------- MLP layer 2 (4 output heads): o = tanh(relu(hid @ w2 + b2)) ----------------
__global__ __launch_bounds__(256) void k_mlp2(const _Float16* __restrict__ hid,
                                              const _Float16* w2q, const _Float16* w2k,
                                              const _Float16* w2v, const _Float16* w2s,
                                              const float* b2q, const float* b2k,
                                              const float* b2v, const float* b2s,
                                              float* __restrict__ o4) {
  int m = blockIdx.x, head = blockIdx.y;
  const _Float16* Wt; const float* bias;
  switch (head) {
    case 0: Wt = w2q; bias = b2q; break;
    case 1: Wt = w2k; bias = b2k; break;
    case 2: Wt = w2v; bias = b2v; break;
    default: Wt = w2s; bias = b2s; break;
  }
  Wt += (size_t)m * Kk * Ff; // rows do (256), ld 1024
  bias += (size_t)m * Kk;
  const _Float16* A = hid + (size_t)(head * Mm + m) * Bb * Ff;
  float* outp = o4 + (size_t)(head * Mm + m) * Bb * Kk;
  int lane = threadIdx.x & 31, w = threadIdx.x >> 5;
  for (int t = w; t < 64; t += 8) {
    int r0 = (t >> 4) * 16, n0 = (t & 15) * 16;
    v8f c = {};
    for (int kk = 0; kk < Ff; kk += 32) {
      v16h a = ldfragA_h(A + (size_t)(r0 + (lane & 15)) * Ff + kk + ((lane >> 4) << 3));
      v16h bb = ldfragB_h(Wt + (size_t)(n0 + (lane & 15)) * Ff + kk + ((lane >> 4) << 3));
      c = wmma16(a, bb, c);
    }
    int col = n0 + (lane & 15);
    int rbase = r0 + ((lane >> 4) << 3);
#pragma unroll
    for (int r = 0; r < 8; ++r)
      outp[(size_t)(rbase + r) * Kk + col] = tanhf(fmaxf(c[r] + bias[col], 0.f));
  }
}

// ---------------- gates[g,m,b] = sigmoid(ghid . g_w2 + g_b2) ----------------
__global__ void k_gates(const _Float16* __restrict__ hid, const float* __restrict__ gw2,
                        const float* __restrict__ gb2, float* __restrict__ gates) {
  int idx = blockIdx.x * 256 + threadIdx.x; // 4*M*B = 2048
  if (idx >= 4 * Mm * Bb) return;
  int b = idx & 63, m = (idx >> 6) & 7, g = idx >> 9;
  const _Float16* hr = hid + (size_t)((4 + g) * Mm + m) * Bb * Ff + (size_t)b * Ff;
  const float* wr = gw2 + (size_t)(g * Mm + m) * Ff;
  float acc = gb2[g * Mm + m];
  for (int f = 0; f < Ff; ++f) acc += (float)hr[f] * wr[f];
  gates[idx] = 1.0f / (1.0f + __expf(-acc));
}

// ---------------- final blend: out_g = gate*o + (1-gate)*x0 ----------------
__global__ void k_final(const float* __restrict__ o4, const float* __restrict__ gates,
                        const float* __restrict__ q0, const float* __restrict__ k0,
                        const float* __restrict__ v0, const float* __restrict__ s0,
                        float* __restrict__ outp) {
  size_t idx = (size_t)blockIdx.x * 256 + threadIdx.x; // 4*M*B*256
  int cc = (int)(idx & 255);
  size_t rest = idx >> 8;
  size_t mb = rest & 511;
  int g = (int)(rest >> 9);
  const float* x0 = (g == 0) ? q0 : (g == 1) ? k0 : (g == 2) ? v0 : s0;
  // d_out regions: attn_out[0], attn_weights[131072], gs[655360], gq[786432], gk[917504], gv[1048576]
  size_t obase = (g == 0) ? 786432u : (g == 1) ? 917504u : (g == 2) ? 1048576u : 655360u;
  float gt = gates[(size_t)g * 512 + mb];
  outp[obase + mb * 256 + cc] = gt * o4[idx] + (1.f - gt) * x0[mb * 256 + cc];
}

extern "C" void kernel_launch(void* const* d_in, const int* in_sizes, int n_in,
                              void* d_out, int out_size, void* d_ws, size_t ws_size,
                              hipStream_t stream) {
  (void)in_sizes; (void)n_in; (void)out_size; (void)ws_size;
  const float* s0v = (const float*)d_in[0];
  const float* q0  = (const float*)d_in[1];
  const float* k0  = (const float*)d_in[2];
  const float* v0  = (const float*)d_in[3];
  const float* key = (const float*)d_in[4];
  const float* val = (const float*)d_in[5];
  const float* wq  = (const float*)d_in[6];
  const float* wk  = (const float*)d_in[7];
  const float* wv  = (const float*)d_in[8];
  const float* bq  = (const float*)d_in[9];
  // d_in[10] = attn_bk: dropped — adds a per-(b,h) constant over s; softmax is shift-invariant.
  const float* bv  = (const float*)d_in[11];
  const float* wo  = (const float*)d_in[12];
  const float* bo  = (const float*)d_in[13];
  const float* w1a[4] = {(const float*)d_in[14], (const float*)d_in[18], (const float*)d_in[22], (const float*)d_in[26]};
  const float* b1a[4] = {(const float*)d_in[15], (const float*)d_in[19], (const float*)d_in[23], (const float*)d_in[27]};
  const float* w2a[4] = {(const float*)d_in[16], (const float*)d_in[20], (const float*)d_in[24], (const float*)d_in[28]};
  const float* b2a[4] = {(const float*)d_in[17], (const float*)d_in[21], (const float*)d_in[25], (const float*)d_in[29]};
  const float* gw1 = (const float*)d_in[30];
  const float* gb1 = (const float*)d_in[31];
  const float* gw2 = (const float*)d_in[32];
  const float* gb2 = (const float*)d_in[33];
  float* outp = (float*)d_out;

  char* wsp = (char*)d_ws;
  size_t off = 0;
  auto alloc = [&](size_t bytes) -> void* {
    void* p = wsp + off;
    off += (bytes + 255) & ~(size_t)255;
    return p;
  };
  _Float16* wqT = (_Float16*)alloc((size_t)Mm * Kk * Kk * 2);
  _Float16* woT = (_Float16*)alloc((size_t)Mm * Kk * Kk * 2);
  _Float16* wvT = (_Float16*)alloc((size_t)Mm * Kk * Kk * 2);
  _Float16* w1T[4]; for (int i = 0; i < 4; ++i) w1T[i] = (_Float16*)alloc((size_t)Mm * Ff * 512 * 2);
  _Float16* gw1T = (_Float16*)alloc((size_t)4 * Mm * Ff * 512 * 2);
  _Float16* w2T[4]; for (int i = 0; i < 4; ++i) w2T[i] = (_Float16*)alloc((size_t)Mm * Kk * Ff * 2);
  _Float16* keyh = (_Float16*)alloc((size_t)Bb * Ss * Kk * 2);
  _Float16* valT = (_Float16*)alloc((size_t)Bb * Kk * Ss * 2);
  float*    qp   = (float*)alloc((size_t)Mm * Bb * Kk * 4);
  _Float16* P    = (_Float16*)alloc((size_t)4 * Bb * 16 * Kk * 2);   // m-paired
  float*    sc   = (float*)alloc((size_t)Mm * Bb * Hh * Ss * 4);
  _Float16* awp  = (_Float16*)alloc((size_t)4 * Bb * 16 * Ss * 2);   // m-paired
  float*    Yb   = (float*)alloc((size_t)Mm * Bb * Hh * Kk * 4);
  _Float16* ctxh = (_Float16*)alloc((size_t)Mm * Bb * Kk * 2);
  _Float16* fcin = (_Float16*)alloc((size_t)Mm * Bb * 512 * 2);
  _Float16* hid  = (_Float16*)alloc((size_t)8 * Mm * Bb * Ff * 2);
  float*    o4   = (float*)alloc((size_t)4 * Mm * Bb * Kk * 4);
  float*    gts  = (float*)alloc((size_t)4 * Mm * Bb * 4);

  // weight transpose + f16 convert (LDS-tiled, coalesced); key/value convert + layout change
  k_tcvt<<<dim3(8, 8, Mm), 256, 0, stream>>>(wq, wqT, 256, 256);
  k_tcvt<<<dim3(8, 8, Mm), 256, 0, stream>>>(wo, woT, 256, 256);
  k_tcvt<<<dim3(8, 8, Mm), 256, 0, stream>>>(wv, wvT, 256, 256);
  for (int i = 0; i < 4; ++i)
    k_tcvt<<<dim3(16, 32, Mm), 256, 0, stream>>>(w1a[i], w1T[i], 512, 1024);
  k_tcvt<<<dim3(16, 32, 4 * Mm), 256, 0, stream>>>(gw1, gw1T, 512, 1024);
  for (int i = 0; i < 4; ++i)
    k_tcvt<<<dim3(32, 8, Mm), 256, 0, stream>>>(w2a[i], w2T[i], 1024, 256);
  k_cvtkey<<<65536, 256, 0, stream>>>(key, keyh);
  k_tval<<<dim3(32, 8, 64), 256, 0, stream>>>(val, valT);

  k_qproj<<<Mm, 256, 0, stream>>>(q0, wqT, bq, qp);
  k_pfold<<<4096, 256, 0, stream>>>(wk, qp, P);
  k_scores<<<dim3(Bb, 4), 256, 0, stream>>>(keyh, P, sc);
  k_softmax<<<dim3(Hh, Bb, Mm), 256, 0, stream>>>(sc, awp);
  k_awmean<<<2048, 256, 0, stream>>>(sc, outp + (size_t)Mm * Bb * Kk);
  k_y<<<dim3(Bb, 4), 256, 0, stream>>>(valT, awp, Yb);
  k_ctx<<<dim3(Hh, Mm), 256, 0, stream>>>(Yb, wvT, bv, ctxh);
  k_attn_out<<<Mm, 256, 0, stream>>>(ctxh, woT, bo, outp, fcin);
  k_fcin_s0<<<512, 256, 0, stream>>>(s0v, fcin);
  k_mlp1<<<dim3(4, Mm, 8), 256, 0, stream>>>(fcin, w1T[0], w1T[1], w1T[2], w1T[3], gw1T,
                                             b1a[0], b1a[1], b1a[2], b1a[3], gb1, hid);
  k_mlp2<<<dim3(Mm, 4), 256, 0, stream>>>(hid, w2T[0], w2T[1], w2T[2], w2T[3],
                                          b2a[0], b2a[1], b2a[2], b2a[3], o4);
  k_gates<<<8, 256, 0, stream>>>(hid, gw2, gb2, gts);
  k_final<<<2048, 256, 0, stream>>>(o4, gts, q0, k0, v0, s0v, outp);
}
